// GAT_ens_2491081032174
// MI455X (gfx1250) — matrix operand order repaired and checked by reading, not compiled, not run on hardware
//
#include <hip/hip_runtime.h>
#include <hip/hip_bf16.h>

// ---------------------------------------------------------------------------
// GAT ensemble for MI455X (gfx1250, wave32).
//  - Dense projections: bf16 WMMA (v_wmma_f32_16x16x32_bf16, fp32 accum),
//    one wave per 16xHDp output strip; A fragment reused across NT tiles;
//    K is compile-time so all B addresses fold into immediate offsets
//    (single base pointer -> no VGPR spills in the WMMA loop).
//  - Edge softmax/aggregation: L2-resident atomics (feat = 51MB << 192MB L2),
//    float4-vectorized gather + atomic scatter.
// ---------------------------------------------------------------------------

typedef __bf16 bf16_t;
typedef __attribute__((ext_vector_type(16))) __bf16 v16bf;
typedef __attribute__((ext_vector_type(8)))  __bf16 v8bf;
typedef __attribute__((ext_vector_type(8)))  float  v8f;

#define SLOPE 0.2f

__device__ __forceinline__ float lrelu(float x) { return x > 0.f ? x : SLOPE * x; }

// Monotonic float<->uint encoding for atomicMax on floats (values may be < 0).
__device__ __forceinline__ unsigned enc_f32(float f) {
    unsigned u = __float_as_uint(f);
    return (u & 0x80000000u) ? ~u : (u | 0x80000000u);
}
__device__ __forceinline__ float dec_f32(unsigned u) {
    return (u & 0x80000000u) ? __uint_as_float(u ^ 0x80000000u) : __uint_as_float(~u);
}
#define ENC_NEG_INF 0x007FFFFFu   // enc(-inf)

// --------------------------- conversion kernels ----------------------------

__global__ void f32_to_bf16_k(const float* __restrict__ in, bf16_t* __restrict__ out,
                              unsigned n) {
    unsigned i = blockIdx.x * blockDim.x + threadIdx.x;
    if (i < n) out[i] = (bf16_t)in[i];
}

// W [K, HD] f32 row-major  ->  Wt [HDpad, K] bf16 (transposed, zero-padded cols)
__global__ void convert_w_k(const float* __restrict__ W, bf16_t* __restrict__ Wt,
                            int K, int HD, int HDpad, unsigned n) {
    unsigned i = blockIdx.x * blockDim.x + threadIdx.x;
    if (i >= n) return;
    int col = (int)(i / (unsigned)K);        // 0..HDpad-1 (output column of GEMM)
    int k   = (int)(i % (unsigned)K);
    Wt[(size_t)col * K + k] = (col < HD) ? (bf16_t)W[(size_t)k * HD + col] : (bf16_t)0.0f;
}

// --------------------------- WMMA GEMM ------------------------------------
// One wave computes a 16 x (NT*16) strip of feat = A(bf16) x Wt^T(bf16).
// A fragment is loaded once per k-step and reused for all NT column tiles.
// K is a template parameter: all B/A fragment addresses become base + constant,
// folding into the 24-bit instruction offset of global_load_b128.
// A: [Nn, K] bf16 row-major.  Wt: [HDpad, K] bf16 (row n = column n of W).
template <int NT, int K>
__global__ void gemm_bf16_wmma_k(const bf16_t* __restrict__ A,
                                 const bf16_t* __restrict__ Wt,
                                 float* __restrict__ C,
                                 int HD, int Nn) {
    const int lane = threadIdx.x;          // 0..31 (wave32)
    const int half = lane >> 4;            // 0 or 1
    const int r    = lane & 15;
    const int tileM = blockIdx.x;

    int arow = tileM * 16 + r;
    if (arow >= Nn) arow = Nn - 1;                 // clamp (Nn % 16 == 0 normally)
    const bf16_t* Arow  = A  + (size_t)arow * K + half * 16;
    const bf16_t* Bbase = Wt + (size_t)r * K + half * 16;

    v8f acc[NT];
    #pragma unroll
    for (int t = 0; t < NT; ++t) acc[t] = {};

    #pragma unroll
    for (int k0 = 0; k0 < K; k0 += 32) {
        // A fragment: elems 0-15 = A[row][k0+16*half ..] (16 contiguous K)
        v8bf alo = *(const v8bf*)(Arow + k0);
        v8bf ahi = *(const v8bf*)(Arow + k0 + 8);
        v16bf av;
        #pragma unroll
        for (int i = 0; i < 8; ++i) { av[i] = alo[i]; av[i + 8] = ahi[i]; }
        // B fragments: lane holds 16 contiguous K values of its column (per tile);
        // t*16*K + k0 is a compile-time constant -> immediate load offset.
        #pragma unroll
        for (int t = 0; t < NT; ++t) {
            v16bf bv = *(const v16bf*)(Bbase + t * 16 * K + k0);
            acc[t] = __builtin_amdgcn_wmma_f32_16x16x32_bf16(
                /*neg_a=*/false, av, /*neg_b=*/false, bv,
                /*c_mod=*/(short)0, acc[t], /*reuse_a=*/false, /*reuse_b=*/false);
        }
    }

    #pragma unroll
    for (int t = 0; t < NT; ++t) {
        const int col = t * 16 + r;
        if (col < HD) {
            #pragma unroll
            for (int i = 0; i < 8; ++i) {
                int row = tileM * 16 + half * 8 + i;
                if (row < Nn) C[(size_t)row * HD + col] = acc[t][i];
            }
        }
    }
}

// --------------------------- attention scores ------------------------------

__global__ void attn_scores_k(const float* __restrict__ feat,
                              const float* __restrict__ al,
                              const float* __restrict__ ar,
                              float* __restrict__ el, float* __restrict__ er,
                              int H, int D, unsigned NH) {
    unsigned i = blockIdx.x * blockDim.x + threadIdx.x;
    if (i >= NH) return;
    int n = (int)(i / (unsigned)H);
    int h = (int)(i % (unsigned)H);
    const float* f = feat + (size_t)n * H * D + (size_t)h * D;
    float sl = 0.f, sr = 0.f;
    for (int d = 0; d < D; ++d) {
        float v = f[d];
        sl += v * al[h * D + d];
        sr += v * ar[h * D + d];
    }
    el[i] = sl;
    er[i] = sr;
}

// --------------------------- accumulator init ------------------------------

__global__ void init_accum_k(unsigned* __restrict__ m_enc, float* __restrict__ den,
                             float* __restrict__ oacc, unsigned NH, unsigned NHD) {
    unsigned i = blockIdx.x * blockDim.x + threadIdx.x;
    if (i < NH) { m_enc[i] = ENC_NEG_INF; den[i] = 0.f; }
    if (i < NHD) oacc[i] = 0.f;
}

// --------------------------- edge phase ------------------------------------

__global__ void edge_max_k(const int* __restrict__ src, const int* __restrict__ dst,
                           const float* __restrict__ el, const float* __restrict__ er,
                           unsigned* __restrict__ m_enc, int E, int H) {
    int e = blockIdx.x * blockDim.x + threadIdx.x;
    if (e >= E) return;
    int s = src[e], d = dst[e];
    for (int h = 0; h < H; ++h) {
        float v = lrelu(el[s * H + h] + er[d * H + h]);
        atomicMax(&m_enc[d * H + h], enc_f32(v));
    }
}

__global__ void edge_exp_k(const int* __restrict__ src, const int* __restrict__ dst,
                           const float* __restrict__ el, const float* __restrict__ er,
                           const unsigned* __restrict__ m_enc,
                           float* __restrict__ den, float* __restrict__ abuf,
                           int E, int H) {
    int e = blockIdx.x * blockDim.x + threadIdx.x;
    if (e >= E) return;
    int s = src[e], d = dst[e];
    for (int h = 0; h < H; ++h) {
        float v = lrelu(el[s * H + h] + er[d * H + h]);
        float a = __expf(v - dec_f32(m_enc[d * H + h]));
        abuf[(size_t)e * H + h] = a;
        atomicAdd(&den[d * H + h], a);
    }
}

// float4 path: one thread per (edge, 4-feature group); requires HD % 4 == 0 and
// D % 4 == 0 so a group never straddles a head boundary.
__global__ void aggregate4_k(const int* __restrict__ src, const int* __restrict__ dst,
                             const float* __restrict__ abuf, const float* __restrict__ feat,
                             float* __restrict__ oacc, int H, int D, int HD,
                             unsigned total4) {
    unsigned i = blockIdx.x * blockDim.x + threadIdx.x;
    if (i >= total4) return;
    const int HD4 = HD >> 2;
    unsigned e = i / (unsigned)HD4;
    int jb = (int)(i - e * (unsigned)HD4) * 4;
    int h = jb / D;
    int s = src[e], d = dst[e];
    float a = abuf[(size_t)e * H + h];
    float4 f = *(const float4*)(feat + (size_t)s * HD + jb);
    float* o = oacc + (size_t)d * HD + jb;
    atomicAdd(o + 0, a * f.x);
    atomicAdd(o + 1, a * f.y);
    atomicAdd(o + 2, a * f.z);
    atomicAdd(o + 3, a * f.w);
}

// scalar fallback (layer 2, HD = 41)
__global__ void aggregate_k(const int* __restrict__ src, const int* __restrict__ dst,
                            const float* __restrict__ abuf, const float* __restrict__ feat,
                            float* __restrict__ oacc, int H, int D, int HD,
                            unsigned total) {
    unsigned i = blockIdx.x * blockDim.x + threadIdx.x;
    if (i >= total) return;
    unsigned e = i / (unsigned)HD;
    int j = (int)(i - e * (unsigned)HD);
    int h = j / D;
    int s = src[e], d = dst[e];
    float a = abuf[(size_t)e * H + h];
    atomicAdd(&oacc[(size_t)d * HD + j], a * feat[(size_t)s * HD + j]);
}

// --------------------------- finalize --------------------------------------

__global__ void finalize_hidden_k(const float* __restrict__ oacc,
                                  const float* __restrict__ den,
                                  bf16_t* __restrict__ h_next,
                                  int H, int D, int HD, unsigned n) {
    unsigned i = blockIdx.x * blockDim.x + threadIdx.x;
    if (i >= n) return;
    unsigned node = i / (unsigned)HD;
    int j = (int)(i - node * (unsigned)HD);
    int h = j / D;
    float v = oacc[i] / fmaxf(den[node * (unsigned)H + h], 1e-9f);
    v = fmaxf(v, 0.f);                          // ReLU
    h_next[i] = (bf16_t)v;                      // bf16 input for next WMMA GEMM
}

__global__ void finalize_last_k(const float* __restrict__ oacc,
                                const float* __restrict__ den,
                                float* __restrict__ out,
                                int H, int D, int HD, unsigned n) {
    unsigned i = blockIdx.x * blockDim.x + threadIdx.x;
    if (i >= n) return;
    unsigned node = i / (unsigned)HD;
    int j = (int)(i - node * (unsigned)HD);
    int h = j / D;
    out[i] = oacc[i] / fmaxf(den[node * (unsigned)H + h], 1e-9f);  // H==1 -> mean==identity
}

// --------------------------- host launcher ---------------------------------

extern "C" void kernel_launch(void* const* d_in, const int* in_sizes, int n_in,
                              void* d_out, int out_size, void* d_ws, size_t ws_size,
                              hipStream_t stream) {
    (void)n_in; (void)out_size; (void)ws_size;
    const float* x   = (const float*)d_in[0];
    const int*   src = (const int*)d_in[1];
    const int*   dst = (const int*)d_in[2];
    const int Nn = in_sizes[0] / 256;   // 100000
    const int E  = in_sizes[1];         // 1600000

    // Workspace carve-out (~168 MB total, 256B-aligned slices).
    char* ws = (char*)d_ws;
    size_t off = 0;
    auto take = [&](size_t bytes) -> char* {
        char* p = ws + off;
        off = (off + bytes + 255) & ~(size_t)255;
        return p;
    };
    bf16_t*   h_bf = (bf16_t*)  take((size_t)Nn * 256 * sizeof(bf16_t));   // activations (bf16)
    bf16_t*   Wt   = (bf16_t*)  take((size_t)128 * 256 * sizeof(bf16_t));  // transposed weights
    float*    feat = (float*)   take((size_t)Nn * 128 * sizeof(float));    // GEMM output
    float*    el   = (float*)   take((size_t)Nn * 2 * sizeof(float));
    float*    er   = (float*)   take((size_t)Nn * 2 * sizeof(float));
    unsigned* menc = (unsigned*)take((size_t)Nn * 2 * sizeof(unsigned));
    float*    den  = (float*)   take((size_t)Nn * 2 * sizeof(float));
    float*    abuf = (float*)   take((size_t)E * 2 * sizeof(float));
    float*    oacc = (float*)   take((size_t)Nn * 128 * sizeof(float));

    const int fans[3] = {256, 128, 128};
    const int Hs[3]   = {2, 2, 1};
    const int Ds[3]   = {64, 64, 41};
    float* out = (float*)d_out;

    for (int b = 0; b < 2; ++b) {
        {   // branch input: x -> bf16
            unsigned n = (unsigned)Nn * 256u;
            f32_to_bf16_k<<<(n + 255) / 256, 256, 0, stream>>>(x, h_bf, n);
        }
        for (int l = 0; l < 3; ++l) {
            const float* W  = (const float*)d_in[3 + b * 9 + l * 3 + 0];
            const float* al = (const float*)d_in[3 + b * 9 + l * 3 + 1];
            const float* ar = (const float*)d_in[3 + b * 9 + l * 3 + 2];
            const int K = fans[l], H = Hs[l], D = Ds[l];
            const int HD = H * D, HDp = (HD + 15) & ~15;

            {   // W -> Wt (bf16, transposed, padded)
                unsigned n = (unsigned)HDp * (unsigned)K;
                convert_w_k<<<(n + 255) / 256, 256, 0, stream>>>(W, Wt, K, HD, HDp, n);
            }
            {   // feat = h @ W via WMMA: one wave per 16-row strip
                int Mtiles = (Nn + 15) / 16;
                if (l == 0) {
                    gemm_bf16_wmma_k<8, 256><<<Mtiles, 32, 0, stream>>>(h_bf, Wt, feat, HD, Nn);
                } else if (l == 1) {
                    gemm_bf16_wmma_k<8, 128><<<Mtiles, 32, 0, stream>>>(h_bf, Wt, feat, HD, Nn);
                } else {  // classifier: HDp == 48
                    gemm_bf16_wmma_k<3, 128><<<Mtiles, 32, 0, stream>>>(h_bf, Wt, feat, HD, Nn);
                }
            }
            {   // el / er
                unsigned n = (unsigned)Nn * (unsigned)H;
                attn_scores_k<<<(n + 255) / 256, 256, 0, stream>>>(feat, al, ar, el, er, H, D, n);
            }
            {   // init accumulators
                unsigned nh = (unsigned)Nn * (unsigned)H, nhd = (unsigned)Nn * (unsigned)HD;
                init_accum_k<<<(nhd + 255) / 256, 256, 0, stream>>>(menc, den, oacc, nh, nhd);
            }
            edge_max_k<<<(E + 255) / 256, 256, 0, stream>>>(src, dst, el, er, menc, E, H);
            edge_exp_k<<<(E + 255) / 256, 256, 0, stream>>>(src, dst, el, er, menc, den, abuf, E, H);
            {   // weighted aggregation (L2-resident atomics)
                if ((HD & 3) == 0 && (D & 3) == 0) {
                    unsigned total4 = (unsigned)E * (unsigned)(HD >> 2);
                    aggregate4_k<<<(total4 + 255) / 256, 256, 0, stream>>>(src, dst, abuf, feat,
                                                                          oacc, H, D, HD, total4);
                } else {
                    unsigned total = (unsigned)E * (unsigned)HD;
                    aggregate_k<<<(total + 255) / 256, 256, 0, stream>>>(src, dst, abuf, feat,
                                                                        oacc, H, D, HD, total);
                }
            }
            if (l < 2) {
                unsigned n = (unsigned)Nn * (unsigned)HD;
                finalize_hidden_k<<<(n + 255) / 256, 256, 0, stream>>>(oacc, den, h_bf, H, D, HD, n);
            } else {
                unsigned n = (unsigned)Nn * (unsigned)HD;   // HD == 41 == CLS
                finalize_last_k<<<(n + 255) / 256, 256, 0, stream>>>(
                    oacc, den, out + (size_t)b * Nn * HD, H, D, HD, n);
            }
        }
    }
}
